// AttentionHead_38938173505637
// MI455X (gfx1250) — matrix operand order repaired and checked by reading, not compiled, not run on hardware
//
#include <hip/hip_runtime.h>

// ---------------------------------------------------------------------------
// Single-head attention, B=4 S=4096 E=1024 H=64, CDNA5 (gfx1250, wave32).
// Pass 0: convert+transpose W -> bf16 (Wq pre-scaled by 1/sqrt(H)=0.125).
// Pass 1: Q/K/V projections with v_wmma_f32_16x16x32_bf16 (f32 accum),
//         B fragments batch-loaded ahead of the 4-WMMA group so loads
//         overlap the XDL pipe instead of serializing on s_wait_loadcnt 0.
// Pass 2: flash attention, one wave per 16-query tile, 32 keys per step,
//         online softmax in C-fragment layout, P via LDS -> A-fragment,
//         K/V fragments batch-loaded, global_prefetch one block ahead.
// ---------------------------------------------------------------------------

typedef __attribute__((ext_vector_type(16))) __bf16 v16bf;
typedef __attribute__((ext_vector_type(8)))  __bf16 v8bf;
typedef __attribute__((ext_vector_type(8)))  float  v8f;

#define S_LEN 4096
#define E_DIM 1024
#define H_DIM 64
#define B_N   4

__device__ __forceinline__ v8f wmma_bf16(v16bf a, v16bf b, v8f c) {
  // 8 args: (neg_a, A, neg_b, B, c_mod, C, reuse_a, reuse_b)
  return __builtin_amdgcn_wmma_f32_16x16x32_bf16(false, a, false, b,
                                                 (short)0, c, false, false);
}

// A-fragment K index for 16-bit A 16x32: lo half lanes carry
// K {0..15}, hi half {16..31}, packed 2 per VGPR (matches B fills).
__device__ __forceinline__ int a_kidx(int i, int hbA) {
  return 2 * hbA + 2 * i;
}

// ---------------------------------------------------------------------------
// Pass 0: Wt[m][h][e] = bf16( W_m[e][h] * (m==0 ? 0.125 : 1) )
// grid = (H*E/256, 3), block = 256
// ---------------------------------------------------------------------------
__global__ void prep_w_kernel(const float* __restrict__ Wq,
                              const float* __restrict__ Wk,
                              const float* __restrict__ Wv,
                              __bf16* __restrict__ Wt) {
  const int m   = blockIdx.y;
  const int idx = blockIdx.x * 256 + threadIdx.x;   // 0 .. H*E-1
  const int h   = idx >> 10;                        // E = 1024
  const int e   = idx & (E_DIM - 1);
  const float* W = (m == 0) ? Wq : ((m == 1) ? Wk : Wv);
  const float s  = (m == 0) ? 0.125f : 1.0f;        // fold 1/sqrt(64) into Wq
  Wt[(size_t)m * (H_DIM * E_DIM) + idx] = (__bf16)(W[e * H_DIM + h] * s);
}

// ---------------------------------------------------------------------------
// Pass 1: projections. grid = (128, 3), block = 256 (8 waves).
// Each wave: one 16-row tile of X times W_mat -> 16x64 tile.
// ---------------------------------------------------------------------------
__global__ void __launch_bounds__(256)
proj_kernel(const float* __restrict__ X, const __bf16* __restrict__ Wt,
            const float* __restrict__ bq, const float* __restrict__ bk,
            const float* __restrict__ bv,
            __bf16* __restrict__ Qb, __bf16* __restrict__ Kb,
            __bf16* __restrict__ Vt) {
  const int lane = threadIdx.x & 31;
  const int wave = threadIdx.x >> 5;
  const int mat  = blockIdx.y;                    // 0=Q 1=K 2=V (uniform)
  const int tile = blockIdx.x * 8 + wave;         // 0..1023
  const int row0 = tile * 16;                     // global row of tile
  const int n_lo = lane & 15;
  const int hiH  = (lane >= 16) ? 1 : 0;
  const int hbA  = hiH ? 8 : 0;
  const int kbB  = hiH ? 16 : 0;

  const __bf16* Wm   = Wt + (size_t)mat * (H_DIM * E_DIM);
  const float*  bias = (mat == 0) ? bq : ((mat == 1) ? bk : bv);
  const float   bsc  = (mat == 0) ? 0.125f : 1.0f;

  v8f acc[4];
#pragma unroll
  for (int t = 0; t < 4; ++t) {
    float bvl = bias[t * 16 + n_lo] * bsc;
#pragma unroll
    for (int r = 0; r < 8; ++r) acc[t][r] = bvl;
  }

  const float* xrow = X + (size_t)(row0 + n_lo) * E_DIM;   // A row = lane&15

  for (int k0 = 0; k0 < E_DIM; k0 += 32) {
    // A fragment: load f32, convert to bf16 inline (v_cvt_pk_bf16_f32).
    v16bf a;
#pragma unroll
    for (int i = 0; i < 8; ++i) {
      int kA = a_kidx(i, hbA);
      float2 f = *(const float2*)(xrow + k0 + kA);         // 8B aligned
      a[2 * i]     = (__bf16)f.x;
      a[2 * i + 1] = (__bf16)f.y;
    }
    // Batch-load ALL four B fragments before any WMMA so the waits are
    // partial and the four independent WMMAs issue back-to-back.
    v16bf bw[4];
#pragma unroll
    for (int t = 0; t < 4; ++t) {
      const __bf16* wrow = Wm + (size_t)(t * 16 + n_lo) * E_DIM + k0 + kbB;
#pragma unroll
      for (int i = 0; i < 8; ++i) {                        // K-contiguous pairs
        bw[t][2 * i]     = wrow[2 * i];
        bw[t][2 * i + 1] = wrow[2 * i + 1];
      }
    }
#pragma unroll
    for (int t = 0; t < 4; ++t) acc[t] = wmma_bf16(a, bw[t], acc[t]);
  }

  // ---- epilogue: single uniform branch on mat, then clean store loops ----
  if (mat == 2) {
    // V transposed: for fixed h, rows r are consecutive s -> one b128 store.
    const int bb = row0 >> 12;
    const int s0 = (row0 & (S_LEN - 1)) + hiH * 8;
#pragma unroll
    for (int t = 0; t < 4; ++t) {
      v8bf pv;
#pragma unroll
      for (int r = 0; r < 8; ++r) pv[r] = (__bf16)acc[t][r];
      __bf16* dst = Vt + ((size_t)bb * H_DIM + t * 16 + n_lo) * S_LEN + s0;
      *(v8bf*)dst = pv;                                    // 16B aligned
    }
  } else {
    __bf16* O = (mat == 0) ? Qb : Kb;
#pragma unroll
    for (int t = 0; t < 4; ++t) {
      int h = t * 16 + n_lo;
#pragma unroll
      for (int r = 0; r < 8; ++r) {
        int row = row0 + r + hiH * 8;                      // C-layout row
        O[(size_t)row * H_DIM + h] = (__bf16)acc[t][r];
      }
    }
  }
}

// ---------------------------------------------------------------------------
// Pass 2: flash attention. grid = (64, 4), block = 128 (4 waves).
// Each wave owns one 16-query tile; loop over keys in blocks of 32.
// ---------------------------------------------------------------------------
__global__ void __launch_bounds__(128)
attn_kernel(const __bf16* __restrict__ Qb, const __bf16* __restrict__ Kb,
            const __bf16* __restrict__ Vt, float* __restrict__ Out) {
  __shared__ __bf16 pbuf[4][16][32];               // 1 KB per wave

  const int lane = threadIdx.x & 31;
  const int wave = threadIdx.x >> 5;
  const int b    = blockIdx.y;
  const int qt   = blockIdx.x * 4 + wave;          // 0..255
  const int q0   = qt * 16;
  const int n_lo = lane & 15;
  const int hiH  = (lane >= 16) ? 1 : 0;
  const int hbA  = hiH ? 8 : 0;
  const int kbB  = hiH ? 16 : 0;

  // Q A-fragments for the two K-steps (h 0..31 and 32..63), scale pre-folded.
  const __bf16* qrow = Qb + ((size_t)b * S_LEN + q0 + n_lo) * H_DIM;
  v16bf qa[2];
#pragma unroll
  for (int ks = 0; ks < 2; ++ks) {
#pragma unroll
    for (int i = 0; i < 8; ++i) {
      int kA = ks * 32 + a_kidx(i, hbA);
      qa[ks][2 * i]     = qrow[kA];
      qa[ks][2 * i + 1] = qrow[kA + 1];
    }
  }

  v8f o[4];
#pragma unroll
  for (int t = 0; t < 4; ++t)
#pragma unroll
    for (int r = 0; r < 8; ++r) o[t][r] = 0.0f;

  float mrow[8], lrow[8];
#pragma unroll
  for (int r = 0; r < 8; ++r) { mrow[r] = -__builtin_inff(); lrow[r] = 0.0f; }

  const __bf16* kbase = Kb + (size_t)b * S_LEN * H_DIM;
  const __bf16* vbase = Vt + (size_t)b * H_DIM * S_LEN;

  for (int j0 = 0; j0 < S_LEN; j0 += 32) {
    // prefetch next key block (gfx1250 global_prefetch_b8); L2-resident K/V.
    if (j0 + 32 < S_LEN) {
      __builtin_prefetch(kbase + (size_t)(j0 + 32 + n_lo) * H_DIM, 0, 0);
      __builtin_prefetch(vbase + (size_t)n_lo * S_LEN + j0 + 32, 0, 0);
    }

    // ---- scores: batch-load all 4 K fragments (2 sub-tiles x 2 K-steps),
    //      then 4 WMMAs (two independent chains of length 2).
    v16bf kf[2][2];
#pragma unroll
    for (int st = 0; st < 2; ++st) {
      const __bf16* krow = kbase + (size_t)(j0 + st * 16 + n_lo) * H_DIM;
#pragma unroll
      for (int ks = 0; ks < 2; ++ks) {
#pragma unroll
        for (int i = 0; i < 8; ++i) {
          int kk = ks * 32 + kbB + 2 * i;          // contiguous along H
          kf[st][ks][2 * i]     = krow[kk];
          kf[st][ks][2 * i + 1] = krow[kk + 1];
        }
      }
    }
    v8f s[2];
#pragma unroll
    for (int st = 0; st < 2; ++st) {
      v8f c;
#pragma unroll
      for (int r = 0; r < 8; ++r) c[r] = 0.0f;
      c = wmma_bf16(qa[0], kf[st][0], c);
      c = wmma_bf16(qa[1], kf[st][1], c);
      s[st] = c;
    }

    // ---- online softmax in C layout (stats per half-wave row group) ----
    float alpha[8];
#pragma unroll
    for (int r = 0; r < 8; ++r) {
      float v = fmaxf(s[0][r], s[1][r]);
      v = fmaxf(v, __shfl_xor(v, 1, 32));
      v = fmaxf(v, __shfl_xor(v, 2, 32));
      v = fmaxf(v, __shfl_xor(v, 4, 32));
      v = fmaxf(v, __shfl_xor(v, 8, 32));          // stays within 16-lane half
      float mnew = fmaxf(mrow[r], v);
      alpha[r]   = __expf(mrow[r] - mnew);
      mrow[r]    = mnew;
      float p0 = __expf(s[0][r] - mnew);
      float p1 = __expf(s[1][r] - mnew);
      s[0][r] = p0; s[1][r] = p1;
      float rs = p0 + p1;
      rs += __shfl_xor(rs, 1, 32);
      rs += __shfl_xor(rs, 2, 32);
      rs += __shfl_xor(rs, 4, 32);
      rs += __shfl_xor(rs, 8, 32);
      lrow[r] = lrow[r] * alpha[r] + rs;
    }

    // ---- rescale running O ----
#pragma unroll
    for (int t = 0; t < 4; ++t)
#pragma unroll
      for (int r = 0; r < 8; ++r) o[t][r] *= alpha[r];

    // ---- P: C layout -> LDS (16x32 bf16, per-wave slice) -> A fragment ----
#pragma unroll
    for (int r = 0; r < 8; ++r) {
      pbuf[wave][hiH * 8 + r][n_lo]      = (__bf16)s[0][r];
      pbuf[wave][hiH * 8 + r][16 + n_lo] = (__bf16)s[1][r];
    }
    v16bf pa;
#pragma unroll
    for (int i = 0; i < 8; ++i) {
      int kA = a_kidx(i, hbA);
      pa[2 * i]     = pbuf[wave][n_lo][kA];
      pa[2 * i + 1] = pbuf[wave][n_lo][kA + 1];
    }

    // ---- O += P (16x32) x V (32 keys x 64 h); batch-load all 4 V frags ----
    v16bf vf[4];
#pragma unroll
    for (int t = 0; t < 4; ++t) {
      const __bf16* vrow = vbase + (size_t)(t * 16 + n_lo) * S_LEN + j0 + kbB;
#pragma unroll
      for (int i = 0; i < 8; ++i) {
        vf[t][2 * i]     = vrow[2 * i];
        vf[t][2 * i + 1] = vrow[2 * i + 1];
      }
    }
#pragma unroll
    for (int t = 0; t < 4; ++t) o[t] = wmma_bf16(pa, vf[t], o[t]);
  }

  // ---- epilogue: normalize (rcp instead of 32 divides) and store f32 ----
  float linv[8];
#pragma unroll
  for (int r = 0; r < 8; ++r) linv[r] = __builtin_amdgcn_rcpf(lrow[r]);
#pragma unroll
  for (int t = 0; t < 4; ++t) {
    int h = t * 16 + n_lo;
#pragma unroll
    for (int r = 0; r < 8; ++r) {
      int tok = q0 + r + hiH * 8;
      Out[((size_t)b * S_LEN + tok) * H_DIM + h] = o[t][r] * linv[r];
    }
  }
}

// ---------------------------------------------------------------------------
extern "C" void kernel_launch(void* const* d_in, const int* in_sizes, int n_in,
                              void* d_out, int out_size, void* d_ws,
                              size_t ws_size, hipStream_t stream) {
  const float* hidden = (const float*)d_in[0];
  const float* Wq     = (const float*)d_in[1];
  const float* bq     = (const float*)d_in[2];
  const float* Wk     = (const float*)d_in[3];
  const float* bk     = (const float*)d_in[4];
  const float* Wv     = (const float*)d_in[5];
  const float* bv     = (const float*)d_in[6];
  float* out = (float*)d_out;

  // workspace layout (bytes): Wt 384KB | Qb 2MB | Kb 2MB | Vt 2MB  (~6.7MB)
  char* ws = (char*)d_ws;
  const size_t WT_BYTES = (size_t)3 * H_DIM * E_DIM * sizeof(__bf16);
  const size_t QB_BYTES = (size_t)B_N * S_LEN * H_DIM * sizeof(__bf16);
  __bf16* Wt = (__bf16*)ws;
  __bf16* Qb = (__bf16*)(ws + WT_BYTES);
  __bf16* Kb = (__bf16*)(ws + WT_BYTES + QB_BYTES);
  __bf16* Vt = (__bf16*)(ws + WT_BYTES + 2 * QB_BYTES);

  dim3 g0(H_DIM * E_DIM / 256, 3);
  prep_w_kernel<<<g0, 256, 0, stream>>>(Wq, Wk, Wv, Wt);

  dim3 g1(128, 3);
  proj_kernel<<<g1, 256, 0, stream>>>(hidden, Wt, bq, bk, bv, Qb, Kb, Vt);

  dim3 g2(64, B_N);
  attn_kernel<<<g2, 128, 0, stream>>>(Qb, Kb, Vt, out);

  (void)in_sizes; (void)n_in; (void)out_size; (void)ws_size;
}
